// GCNWithResidual_43052752175807
// MI455X (gfx1250) — matrix operand order repaired and checked by reading, not compiled, not run on hardware
//
#include <hip/hip_runtime.h>
#include <hip/hip_bf16.h>

typedef __attribute__((ext_vector_type(16))) __bf16 v16bf;
typedef __attribute__((ext_vector_type(8)))  __bf16 v8bf;
typedef __attribute__((ext_vector_type(4)))  __bf16 v4bf;
typedef __attribute__((ext_vector_type(8)))  float  v8f;

#define D 128
#define LN_EPS 1e-5f

// ---------------- elementwise / setup kernels ----------------

__global__ __launch_bounds__(256) void k_convert_x(const float* __restrict__ x,
                                                   __bf16* __restrict__ xb, int n) {
  int i = blockIdx.x * blockDim.x + threadIdx.x;
  if (i < n) xb[i] = (__bf16)x[i];
}

// Pack weights transposed: wt[((l*2+m)*128 + n)*128 + k] = W_m[l][k][n], bf16.
__global__ __launch_bounds__(256) void k_convert_w(const float* __restrict__ conv_w,
                                                   const float* __restrict__ res_w,
                                                   __bf16* __restrict__ wt, int total) {
  int idx = blockIdx.x * blockDim.x + threadIdx.x;
  if (idx >= total) return;
  int k = idx & 127;
  int n = (idx >> 7) & 127;
  int m = (idx >> 14) & 1;
  int l = idx >> 15;
  const float* s = m ? res_w : conv_w;
  wt[idx] = (__bf16)s[(size_t)l * D * D + (size_t)k * D + n];
}

__global__ __launch_bounds__(256) void k_zero(float* __restrict__ p, int n) {
  int i = blockIdx.x * blockDim.x + threadIdx.x;
  if (i < n) p[i] = 0.0f;
}

__global__ __launch_bounds__(256) void k_deg(const int* __restrict__ ei,
                                             float* __restrict__ deg, int E) {
  int e = blockIdx.x * blockDim.x + threadIdx.x;
  if (e < E) atomicAdd(&deg[ei[E + e]], 1.0f);
}

// deg -> dinv in place; self-loop weight 2.0 folded in (deg always > 0).
__global__ __launch_bounds__(256) void k_dinv(float* __restrict__ d, int n) {
  int i = blockIdx.x * blockDim.x + threadIdx.x;
  if (i < n) d[i] = rsqrtf(d[i] + 2.0f);
}

// ---------------- WMMA GEMM: C[N,128] = A[N,128] * B[128,128] ----------------
// A: bf16 row-major.  Bt: bf16, Bt[n][k] = B[k][n] (pre-transposed so each lane's
// B fragment, like the A fragment, is two contiguous 16B loads,
// matching the ISA 16-bit A 16x32 / B 32x16 VGPR layouts).  One wave = one 16x16
// C tile; 8 waves/block cover all 128 output columns of one 16-row tile;
// K=128 -> 4 x v_wmma_f32_16x16x32_bf16.
//
// Optional fusion: when agg != nullptr, also emit
//   agg[i][f] = conv_b[f] + 2*dinv[i]^2 * C[i][f]
// (GCN self-loop with improved=True weight 2.0) straight from the accumulator,
// saving a full re-read of C.
__global__ __launch_bounds__(256) void k_gemm(const __bf16* __restrict__ A,
                                              const __bf16* __restrict__ Bt,
                                              float* __restrict__ C, int N,
                                              const float* __restrict__ dinv,
                                              const float* __restrict__ conv_b,
                                              float* __restrict__ agg) {
  const int lane = threadIdx.x & 31;
  const int wave = threadIdx.x >> 5;           // column tile 0..7
  const int rt   = blockIdx.x;                 // row tile
  int row = rt * 16 + (lane & 15);
  if (row >= N) row = N - 1;                   // clamp (N==50000 is tile-exact)
  const int col = wave * 16 + (lane & 15);
  const int kbA = (lane >> 4) * 8;             // A: lanes 16-31 hold K+8
  const int kbB = (lane >> 4) * 8;             // B: lanes 16-31 hold K+8
  const __bf16* ar = A  + (size_t)row * D;
  const __bf16* bc = Bt + (size_t)col * D;

  v8f acc = {};
#pragma unroll
  for (int kc = 0; kc < 4; ++kc) {
    const int k0 = kc * 32;
    v8bf alo = *(const v8bf*)(ar + k0 + kbA);        // K = k0+kb .. +7
    v8bf ahi = *(const v8bf*)(ar + k0 + 16 + kbA);   // K = k0+16+kb .. +7
    v16bf a = __builtin_shufflevector(alo, ahi,
        0, 1, 2, 3, 4, 5, 6, 7, 8, 9, 10, 11, 12, 13, 14, 15);
    v8bf blo = *(const v8bf*)(bc + k0 + kbB);        // K = k0+kb .. +7
    v8bf bhi = *(const v8bf*)(bc + k0 + 16 + kbB);   // K = k0+16+kb .. +7
    v16bf b = __builtin_shufflevector(blo, bhi,
        0, 1, 2, 3, 4, 5, 6, 7, 8, 9, 10, 11, 12, 13, 14, 15);
    acc = __builtin_amdgcn_wmma_f32_16x16x32_bf16(
        false, a, false, b, (short)0, acc, false, false);
  }

  const int crow = rt * 16 + (lane >> 4) * 8;        // C layout: VGPR v -> row crow+v
  float* cp = C + (size_t)crow * D + col;
  if (rt * 16 + 16 <= N) {                           // uniform whole-tile fast path
#pragma unroll
    for (int v = 0; v < 8; ++v) cp[(size_t)v * D] = acc[v];
    if (agg) {
      const float cb = conv_b[col];
      float* ap = agg + (size_t)crow * D + col;
#pragma unroll
      for (int v = 0; v < 8; ++v) {
        float di = dinv[crow + v];
        ap[(size_t)v * D] = cb + 2.0f * di * di * acc[v];
      }
    }
  } else {                                           // ragged tail (unused for N=50000)
    const float cb = agg ? conv_b[col] : 0.0f;
    for (int v = 0; v < 8; ++v) {
      if (crow + v < N) {
        cp[(size_t)v * D] = acc[v];
        if (agg) {
          float di = dinv[crow + v];
          agg[(size_t)(crow + v) * D + col] = cb + 2.0f * di * di * acc[v];
        }
      }
    }
  }
}

// ---------------- message passing ----------------

// One wave per edge, 4 features per lane; f32 atomics land in L2 (hw/agg both
// L2-resident at 25.6 MB each, well under the 192 MB global L2).
__global__ __launch_bounds__(256) void k_edge_scatter(const float* __restrict__ hw,
                                                      const int* __restrict__ ei,
                                                      const float* __restrict__ dinv,
                                                      float* __restrict__ agg, int E) {
  int e = blockIdx.x * (blockDim.x >> 5) + (threadIdx.x >> 5);
  if (e >= E) return;
  int lane = threadIdx.x & 31;
  int r = ei[e];
  int c = ei[E + e];
  float norm = dinv[r] * dinv[c];
  float4 v = ((const float4*)(hw + (size_t)r * D))[lane];
  float* dst = agg + (size_t)c * D + lane * 4;
  atomicAdd(dst + 0, v.x * norm);
  atomicAdd(dst + 1, v.y * norm);
  atomicAdd(dst + 2, v.z * norm);
  atomicAdd(dst + 3, v.w * norm);
}

// ---------------- BatchNorm column stats ----------------
// stride (grid*block) is a multiple of 128, so each thread owns one feature.
__global__ __launch_bounds__(256) void k_bn_stats(const float* __restrict__ agg,
                                                  float* __restrict__ gsum,  // [0:128) sum, [128:256) sumsq
                                                  int N) {
  int tid = blockIdx.x * blockDim.x + threadIdx.x;
  int total = gridDim.x * blockDim.x;
  int f = threadIdx.x & 127;
  float s = 0.0f, q = 0.0f;
  size_t nd = (size_t)N * D;
  for (size_t i = tid; i < nd; i += total) {
    float v = agg[i];
    s += v;
    q += v * v;
  }
  __shared__ float ss[256], sq[256];
  ss[threadIdx.x] = s;
  sq[threadIdx.x] = q;
  __syncthreads();
  if (threadIdx.x < 128) {
    atomicAdd(&gsum[f],       ss[threadIdx.x] + ss[threadIdx.x + 128]);
    atomicAdd(&gsum[128 + f], sq[threadIdx.x] + sq[threadIdx.x + 128]);
  }
}

// scale/shift per feature: out = agg*scale + shift implements BN affine.
__global__ __launch_bounds__(128) void k_bn_scale(const float* __restrict__ gsum,
                                                  const float* __restrict__ bn_g,
                                                  const float* __restrict__ bn_b,
                                                  float* __restrict__ ss, int N) {
  int f = threadIdx.x;
  float invN = 1.0f / (float)N;
  float mean = gsum[f] * invN;
  float var  = gsum[128 + f] * invN - mean * mean;
  float inv  = rsqrtf(var + LN_EPS);
  float sc   = inv * bn_g[f];
  ss[f]       = sc;
  ss[128 + f] = bn_b[f] - mean * sc;
}

// ---------------- fused BN-apply + residual + ReLU + LayerNorm ----------------
// One wave per row (wave32), 4 features/lane, shuffle reductions for LN.
__global__ __launch_bounds__(256) void k_finalize(const float* __restrict__ agg,
                                                  const float* __restrict__ rp,
                                                  const float* __restrict__ ss,
                                                  const float* __restrict__ res_b,
                                                  const float* __restrict__ ln_g,
                                                  const float* __restrict__ ln_b,
                                                  __bf16* __restrict__ hb,
                                                  float* __restrict__ hout,
                                                  int N, int wout) {
  int lane = threadIdx.x & 31;
  int row  = blockIdx.x * 8 + (threadIdx.x >> 5);
  if (row >= N) return;
  size_t base = (size_t)row * D;

  float4 av = ((const float4*)(agg + base))[lane];
  float4 rv = ((const float4*)(rp + base))[lane];
  float4 sc = ((const float4*)ss)[lane];
  float4 sh = ((const float4*)(ss + 128))[lane];
  float4 rb = ((const float4*)res_b)[lane];

  float o0 = fmaxf(av.x * sc.x + sh.x + rv.x + rb.x, 0.0f);
  float o1 = fmaxf(av.y * sc.y + sh.y + rv.y + rb.y, 0.0f);
  float o2 = fmaxf(av.z * sc.z + sh.z + rv.z + rb.z, 0.0f);
  float o3 = fmaxf(av.w * sc.w + sh.w + rv.w + rb.w, 0.0f);

  float s = o0 + o1 + o2 + o3;
#pragma unroll
  for (int m = 16; m >= 1; m >>= 1) s += __shfl_xor(s, m, 32);
  float mean = s * (1.0f / 128.0f);

  float d0 = o0 - mean, d1 = o1 - mean, d2 = o2 - mean, d3 = o3 - mean;
  float q = d0 * d0 + d1 * d1 + d2 * d2 + d3 * d3;
#pragma unroll
  for (int m = 16; m >= 1; m >>= 1) q += __shfl_xor(q, m, 32);
  float inv = rsqrtf(q * (1.0f / 128.0f) + LN_EPS);

  float4 lg = ((const float4*)ln_g)[lane];
  float4 lb = ((const float4*)ln_b)[lane];
  float r0 = d0 * inv * lg.x + lb.x;
  float r1 = d1 * inv * lg.y + lb.y;
  float r2 = d2 * inv * lg.z + lb.z;
  float r3 = d3 * inv * lg.w + lb.w;

  v4bf pk = { (__bf16)r0, (__bf16)r1, (__bf16)r2, (__bf16)r3 };
  *(v4bf*)(hb + base + lane * 4) = pk;
  if (wout) {
    float4 o = { r0, r1, r2, r3 };
    ((float4*)(hout + base))[lane] = o;
  }
}

// ---------------- driver ----------------

extern "C" void kernel_launch(void* const* d_in, const int* in_sizes, int n_in,
                              void* d_out, int out_size, void* d_ws, size_t ws_size,
                              hipStream_t stream) {
  const float* x      = (const float*)d_in[0];
  const int*   ei     = (const int*)d_in[1];
  const float* conv_w = (const float*)d_in[2];
  const float* conv_b = (const float*)d_in[3];
  const float* bn_g   = (const float*)d_in[4];
  const float* bn_b   = (const float*)d_in[5];
  const float* res_w  = (const float*)d_in[6];
  const float* res_b  = (const float*)d_in[7];
  const float* ln_g   = (const float*)d_in[8];
  const float* ln_b   = (const float*)d_in[9];

  const int N = in_sizes[0] / D;
  const int E = in_sizes[1] / 2;
  const int L = in_sizes[2] / (D * D);
  const int ND = N * D;

  // workspace layout (256B aligned slices)
  char* ws = (char*)d_ws;
  size_t off = 0;
  auto carve = [&](size_t bytes) -> char* {
    char* p = ws + off;
    off += (bytes + 255) & ~(size_t)255;
    return p;
  };
  float*  hw   = (float*)carve((size_t)ND * 4);       // conv GEMM out, reused for res GEMM
  float*  agg  = (float*)carve((size_t)ND * 4);       // aggregation
  __bf16* hb   = (__bf16*)carve((size_t)ND * 2);      // bf16 copy of h
  float*  dinv = (float*)carve((size_t)N * 4);        // deg -> d^{-1/2}
  __bf16* wt   = (__bf16*)carve((size_t)L * 2 * D * D * 2); // packed transposed weights
  float*  gsum = (float*)carve(256 * 4);              // BN sum / sumsq
  float*  ss   = (float*)carve(256 * 4);              // BN scale / shift
  (void)ws_size; (void)n_in; (void)out_size;

  const int B = 256;
  k_convert_x<<<(ND + B - 1) / B, B, 0, stream>>>(x, hb, ND);
  const int WT = L * 2 * D * D;
  k_convert_w<<<(WT + B - 1) / B, B, 0, stream>>>(conv_w, res_w, wt, WT);
  k_zero<<<(N + B - 1) / B, B, 0, stream>>>(dinv, N);
  k_deg<<<(E + B - 1) / B, B, 0, stream>>>(ei, dinv, E);
  k_dinv<<<(N + B - 1) / B, B, 0, stream>>>(dinv, N);

  const int gemmGrid = (N + 15) / 16;
  for (int l = 0; l < L; ++l) {
    const __bf16* wc = wt + (size_t)(l * 2 + 0) * D * D;
    const __bf16* wr = wt + (size_t)(l * 2 + 1) * D * D;

    // hw = h @ Wc, fused with agg = conv_b + self-loop contribution
    k_gemm<<<gemmGrid, B, 0, stream>>>(hb, wc, hw, N, dinv, conv_b + l * D, agg);
    k_edge_scatter<<<(E + 7) / 8, B, 0, stream>>>(hw, ei, dinv, agg, E);
    k_zero<<<1, B, 0, stream>>>(gsum, 256);
    k_bn_stats<<<512, B, 0, stream>>>(agg, gsum, N);
    k_bn_scale<<<1, 128, 0, stream>>>(gsum, bn_g + l * D, bn_b + l * D, ss, N);
    // hw = h @ Wr (buffer reuse; safe, stream-ordered after scatter/stats)
    k_gemm<<<gemmGrid, B, 0, stream>>>(hb, wr, hw, N, nullptr, nullptr, nullptr);
    k_finalize<<<(N + 7) / 8, B, 0, stream>>>(agg, hw, ss, res_b + l * D,
                                              ln_g, ln_b, hb, (float*)d_out,
                                              N, l == L - 1 ? 1 : 0);
  }
}
